// predictor_5875515261324
// MI455X (gfx1250) — matrix operand order repaired and checked by reading, not compiled, not run on hardware
//
#include <hip/hip_runtime.h>
#include <hip/hip_bf16.h>
#include <stddef.h>

// ---------------------------------------------------------------------------
// Types for WMMA fragments (trivial ext-vector types so unions are legal)
// ---------------------------------------------------------------------------
typedef __attribute__((ext_vector_type(16))) __bf16   v16bf;
typedef __attribute__((ext_vector_type(8)))  float    v8f;
typedef __attribute__((ext_vector_type(4)))  unsigned u32x4;

union FragU {            // 32 bytes = 16 bf16 = one WMMA A/B fragment per lane
    u32x4 q[2];
    v16bf v;
};

__device__ __forceinline__ unsigned short f2bf(float f) {
    unsigned u = __float_as_uint(f);
    unsigned r = u + 0x7FFFu + ((u >> 16) & 1u);   // round-to-nearest-even
    return (unsigned short)(r >> 16);
}

__device__ __forceinline__ float sigm(float x) {
    return 1.0f / (1.0f + __expf(-x));
}
__device__ __forceinline__ float tanh_f(float x) {
    return 2.0f * sigm(2.0f * x) - 1.0f;
}

// Single opaque 32-bit value per timestep: everything derived from it is
// loop-variant as far as LLVM knows, so weight loads cannot be hoisted out
// of the time loop (which previously caused scratch spills). Keeping ONE
// opaque value (instead of per-fragment opaque offsets) means there is no
// large invariant set to spill: per-fragment addressing folds into the
// global_load_b128 immediate offset field.
__device__ __forceinline__ unsigned opaque_zero() {
    unsigned z = 0;
    asm volatile("" : "+v"(z));
    return z;
}

#define HID   256
#define GATES 1024            // 4*HID
#define KCAT  512             // [h1 | h2]

// ---------------------------------------------------------------------------
// Prep kernel: fp32 -> bf16 weight repack + fused biases
//   wb1  [1024][256] bf16 : W_hh1 row-major
//   wbc  [1024][512] bf16 : [W_ih2 | W_hh2] row-major (K-concat for layer 2)
//   b1   [1024] f32       : b_ih1 + b_hh1
//   b2   [1024] f32       : b_ih2 + b_hh2
//   wcol [1024] f32       : W_ih1[:,0]  (layer-1 scalar-input column)
// ---------------------------------------------------------------------------
__global__ __launch_bounds__(256) void lstm_prep(
    const float* __restrict__ Wih1, const float* __restrict__ Whh1,
    const float* __restrict__ bih1, const float* __restrict__ bhh1,
    const float* __restrict__ Wih2, const float* __restrict__ Whh2,
    const float* __restrict__ bih2, const float* __restrict__ bhh2,
    unsigned short* __restrict__ wb1, unsigned short* __restrict__ wbc,
    float* __restrict__ b1, float* __restrict__ b2, float* __restrict__ wcol)
{
    int i = blockIdx.x * blockDim.x + threadIdx.x;
    if (i < GATES * HID)
        wb1[i] = f2bf(Whh1[i]);
    if (i < GATES * KCAT) {
        int j = i >> 9;            // gate row
        int c = i & 511;           // K within [h1|h2]
        float v = (c < HID) ? Wih2[j * HID + c] : Whh2[j * HID + (c - HID)];
        wbc[i] = f2bf(v);
    }
    if (i < GATES) {
        b1[i]   = bih1[i] + bhh1[i];
        b2[i]   = bih2[i] + bhh2[i];
        wcol[i] = Wih1[i];
    }
}

// ---------------------------------------------------------------------------
// Persistent 2-layer LSTM kernel.
//  - 1 workgroup = 16 samples, 8 waves (256 threads), runs all T steps.
//  - Wave w owns gate tiles {w, w+8} inside EACH of the i/f/g/o blocks, so
//    i,f,g,o for a given (sample, hidden unit) land in the same lane/VGPR:
//    the cell update is fully register-resident. c1/c2 live in registers.
//  - h1|h2 kept in LDS as bf16 [16][512]; WMMA A fragments are ds_load_b128.
//  - Weights stream from L2 as global_load_b128 directly into B fragments,
//    re-read every step (1.5MB shared set, L2-resident on a 192MB L2).
// ---------------------------------------------------------------------------
__global__ __launch_bounds__(256) void lstm_persist(
    const float* __restrict__ x,          // [N, L]
    const float* __restrict__ Wl,         // [256]
    const float* __restrict__ blp,        // [1]
    const unsigned short* __restrict__ wb1,   // [1024][256] bf16
    const unsigned short* __restrict__ wbc,   // [1024][512] bf16
    const float* __restrict__ b1g,        // [1024]
    const float* __restrict__ b2g,        // [1024]
    const float* __restrict__ wcolg,      // [1024]
    float* __restrict__ out,              // [N, T]
    int L, int T)
{
    __shared__ __align__(16) unsigned short hcat[16][KCAT]; // [sample][h1|h2] bf16
    __shared__ float xbuf[16];
    __shared__ float red[16][8];
    __shared__ float outbuf[16];

    const int tid = threadIdx.x;
    const int w   = tid >> 5;          // wave 0..7
    const int l   = tid & 31;          // lane
    const int col = l & 15;            // WMMA N-column / A-row
    const int hi  = l >> 4;            // lane half
    const int n0  = blockIdx.x * 16;   // batch tile base

    // zero initial hidden state
    for (int i = tid; i < 16 * KCAT; i += 256)
        ((unsigned short*)hcat)[i] = 0;

    // per-lane constants: hidden units this lane owns: u(s) = (w + 8s)*16 + col
    float b1r[4][2], b2r[4][2], xwr[4][2], wlr[2];
    int   uu[2];
#pragma unroll
    for (int s = 0; s < 2; ++s) {
        int u = (w + s * 8) * 16 + col;
        uu[s]  = u;
        wlr[s] = Wl[u];
#pragma unroll
        for (int b = 0; b < 4; ++b) {
            b1r[b][s] = b1g[b * HID + u];
            b2r[b][s] = b2g[b * HID + u];
            xwr[b][s] = wcolg[b * HID + u];
        }
    }
    const float blv = blp[0];

    // cell states in registers: [s][vgpr] ; sample m = v + 8*hi, unit = uu[s]
    float c1r[2][8], c2r[2][8];
#pragma unroll
    for (int s = 0; s < 2; ++s)
#pragma unroll
        for (int v = 0; v < 8; ++v) { c1r[s][v] = 0.f; c2r[s][v] = 0.f; }

    const v8f zero8 = {0.f, 0.f, 0.f, 0.f, 0.f, 0.f, 0.f, 0.f};

    for (int t = 0; t < T; ++t) {
        if (tid < 16)
            xbuf[tid] = (t < L) ? x[(size_t)(n0 + tid) * L + t] : outbuf[tid];
        __syncthreads();   // xbuf ready; prev-step hcat writes visible

        // One opaque value per step: all weight addresses derive from it.
        const unsigned zo = opaque_zero();

        // ================= layer 1: gates1 = h1 @ W_hh1^T (K=256) ==========
        v8f acc[4][2];
#pragma unroll
        for (int b = 0; b < 4; ++b)
#pragma unroll
            for (int s = 0; s < 2; ++s) acc[b][s] = zero8;

        // 8 row-base pointers (recomputed each step; K-step offsets fold
        // into the load-instruction immediate field)
        const unsigned short* pb1[4][2];
#pragma unroll
        for (int b = 0; b < 4; ++b)
#pragma unroll
            for (int s = 0; s < 2; ++s) {
                const int j = b * HID + (w + s * 8) * 16 + col;
                pb1[b][s] = wb1 + (unsigned)(j * HID + hi * 8) + zo;
            }

#pragma unroll
        for (int kk = 0; kk < 8; ++kk) {
            const int k0 = kk * 32;
            FragU a;   // A: lane row = sample 'col'; K chunks per ISA layout
            a.q[0] = *(const u32x4*)&hcat[col][k0 +      hi * 8];
            a.q[1] = *(const u32x4*)&hcat[col][k0 + 16 + hi * 8];
#pragma unroll
            for (int b = 0; b < 4; ++b) {
#pragma unroll
                for (int s = 0; s < 2; ++s) {
                    const unsigned short* bp = pb1[b][s] + k0;   // imm offset
                    FragU bb;
                    bb.q[0] = *(const u32x4*)(bp);
                    bb.q[1] = *(const u32x4*)(bp + 16);
                    acc[b][s] = __builtin_amdgcn_wmma_f32_16x16x32_bf16(
                        false, a.v, false, bb.v, (short)0, acc[b][s], false, false);
                }
            }
        }

        // in-register gate nonlinearities + c1/h1 update
#pragma unroll
        for (int s = 0; s < 2; ++s) {
#pragma unroll
            for (int v = 0; v < 8; ++v) {
                const int m  = v + 8 * hi;
                const float xm = xbuf[m];
                float gi = acc[0][s][v] + b1r[0][s] + xm * xwr[0][s];
                float gf = acc[1][s][v] + b1r[1][s] + xm * xwr[1][s];
                float gg = acc[2][s][v] + b1r[2][s] + xm * xwr[2][s];
                float go = acc[3][s][v] + b1r[3][s] + xm * xwr[3][s];
                float i_ = sigm(gi), f_ = sigm(gf);
                float g_ = tanh_f(gg), o_ = sigm(go);
                float cn = f_ * c1r[s][v] + i_ * g_;
                c1r[s][v] = cn;
                hcat[m][uu[s]] = f2bf(o_ * tanh_f(cn));   // h1 -> LDS (bf16)
            }
        }
        __syncthreads();   // h1 complete for all waves

        // ========== layer 2: gates2 = [h1|h2] @ [W_ih2|W_hh2]^T (K=512) ====
#pragma unroll
        for (int b = 0; b < 4; ++b)
#pragma unroll
            for (int s = 0; s < 2; ++s) acc[b][s] = zero8;

        const unsigned short* pb2[4][2];
#pragma unroll
        for (int b = 0; b < 4; ++b)
#pragma unroll
            for (int s = 0; s < 2; ++s) {
                const int j = b * HID + (w + s * 8) * 16 + col;
                pb2[b][s] = wbc + (unsigned)(j * KCAT + hi * 8) + zo;
            }

#pragma unroll
        for (int kk = 0; kk < 16; ++kk) {
            const int k0 = kk * 32;
            FragU a;
            a.q[0] = *(const u32x4*)&hcat[col][k0 +      hi * 8];
            a.q[1] = *(const u32x4*)&hcat[col][k0 + 16 + hi * 8];
#pragma unroll
            for (int b = 0; b < 4; ++b) {
#pragma unroll
                for (int s = 0; s < 2; ++s) {
                    const unsigned short* bp = pb2[b][s] + k0;   // imm offset
                    FragU bb;
                    bb.q[0] = *(const u32x4*)(bp);
                    bb.q[1] = *(const u32x4*)(bp + 16);
                    acc[b][s] = __builtin_amdgcn_wmma_f32_16x16x32_bf16(
                        false, a.v, false, bb.v, (short)0, acc[b][s], false, false);
                }
            }
        }

        // c2/h2 update + output-projection partials
        float ps[8];
#pragma unroll
        for (int v = 0; v < 8; ++v) ps[v] = 0.f;

#pragma unroll
        for (int s = 0; s < 2; ++s) {
#pragma unroll
            for (int v = 0; v < 8; ++v) {
                const int m = v + 8 * hi;
                float gi = acc[0][s][v] + b2r[0][s];
                float gf = acc[1][s][v] + b2r[1][s];
                float gg = acc[2][s][v] + b2r[2][s];
                float go = acc[3][s][v] + b2r[3][s];
                float i_ = sigm(gi), f_ = sigm(gf);
                float g_ = tanh_f(gg), o_ = sigm(go);
                float cn = f_ * c2r[s][v] + i_ * g_;
                c2r[s][v] = cn;
                float h = o_ * tanh_f(cn);
                hcat[m][HID + uu[s]] = f2bf(h);           // h2 -> LDS (bf16)
                ps[v] += h * wlr[s];                      // h2 . Wl partial
            }
        }

        // reduce ps over the 16-lane group (covers 16 hidden-unit columns)
#pragma unroll
        for (int v = 0; v < 8; ++v) {
            float p = ps[v];
#pragma unroll
            for (int off = 1; off < 16; off <<= 1)
                p += __shfl_xor(p, off, 32);
            ps[v] = p;
        }
        if (col == 0) {
#pragma unroll
            for (int v = 0; v < 8; ++v)
                red[v + 8 * hi][w] = ps[v];
        }
        __syncthreads();

        if (tid < 16) {
            float o = blv;
#pragma unroll
            for (int ww = 0; ww < 8; ++ww) o += red[tid][ww];
            outbuf[tid] = o;                              // feeds future steps
            out[(size_t)(n0 + tid) * T + t] = o;
        }
        __syncthreads();
    }
}

// ---------------------------------------------------------------------------
// Host launcher
// ---------------------------------------------------------------------------
extern "C" void kernel_launch(void* const* d_in, const int* in_sizes, int n_in,
                              void* d_out, int out_size, void* d_ws, size_t ws_size,
                              hipStream_t stream)
{
    const float* x    = (const float*)d_in[0];
    const float* Wih1 = (const float*)d_in[1];
    const float* Whh1 = (const float*)d_in[2];
    const float* bih1 = (const float*)d_in[3];
    const float* bhh1 = (const float*)d_in[4];
    const float* Wih2 = (const float*)d_in[5];
    const float* Whh2 = (const float*)d_in[6];
    const float* bih2 = (const float*)d_in[7];
    const float* bhh2 = (const float*)d_in[8];
    const float* Wl   = (const float*)d_in[9];
    const float* bl   = (const float*)d_in[10];

    const int N = 4096;
    const int L = in_sizes[0] / N;       // 512
    const int T = out_size  / N;         // 576 = L + future

    // workspace layout
    char* ws = (char*)d_ws;
    unsigned short* wb1  = (unsigned short*)(ws);                   // 512 KB
    unsigned short* wbc  = (unsigned short*)(ws + 512 * 1024);      // 1 MB
    float*          b1   = (float*)(ws + 1536 * 1024);              // 4 KB
    float*          b2   = b1 + GATES;                              // 4 KB
    float*          wcol = b2 + GATES;                              // 4 KB

    // weight repack (bf16) + fused biases
    lstm_prep<<<(GATES * KCAT + 255) / 256, 256, 0, stream>>>(
        Wih1, Whh1, bih1, bhh1, Wih2, Whh2, bih2, bhh2,
        wb1, wbc, b1, b2, wcol);

    // persistent LSTM: 256 workgroups x 16 samples, 8 waves each
    lstm_persist<<<N / 16, 256, 0, stream>>>(
        x, Wl, bl, wb1, wbc, b1, b2, wcol, (float*)d_out, L, T);
}
